// TransformerDecoder_1116691497780
// MI455X (gfx1250) — matrix operand order repaired and checked by reading, not compiled, not run on hardware
//
#include <hip/hip_runtime.h>

// ---------------------------------------------------------------------------
// WLN message-passing network on gfx1250 using v_wmma_f32_16x16x32_bf16.
// All WMMA operands are kept in bf16 mirrors in d_ws, converted exactly once
// at their producer; f32 output is written only on the final layer.
//
//  prep:     tgt -> vf_bf16 ; (bond_adj->edge->bond_table) -> edge_pack ;
//            weights -> bf16, zero-padded to 32-wide K-chunks
//  Layer i:
//    kernel A: nei_bf16[bv][d] = bf16( sum_k mask[k] *
//                 lrelu( l_nei[bv][k][:] . U2_w[d][:] + U2_b[d] ) )
//              (one wave per (b,v); K=16 neighbors == WMMA M dimension)
//    kernel B: vf = lrelu([vf|nei] . U1_w^T + U1_b): writes bf16 mirror
//              in-place (+ f32 d_out on last layer). One wave per 16-row tile.
// ---------------------------------------------------------------------------

typedef __attribute__((ext_vector_type(16))) __bf16 v16bf;
typedef __attribute__((ext_vector_type(8)))  float  v8f;

union BfVec { __bf16 e[16]; v16bf v; uint4 q[2]; };
union Bf8   { __bf16 e[8];  uint4 q; };

__device__ inline __bf16 bfzero() {
  unsigned short z = 0;
  return __builtin_bit_cast(__bf16, z);
}

// fp32 -> bf16 round-to-nearest-even
__device__ inline __bf16 f2bf(float f) {
  unsigned u = __builtin_bit_cast(unsigned, f);
  u += 0x7FFFu + ((u >> 16) & 1u);
  unsigned short h = (unsigned short)(u >> 16);
  return __builtin_bit_cast(__bf16, h);
}

// leaky_relu(x, 0.1) == max(x, 0.1*x): one v_mul + one v_max_num_f32
__device__ inline float lrelu(float x) { return fmaxf(x, 0.1f * x); }

// 16 consecutive bf16 (16B-aligned) -> v16bf
__device__ inline v16bf load_bf16x16(const __bf16* p) {
  BfVec r;
  r.q[0] = *(const uint4*)(p);
  r.q[1] = *(const uint4*)(p + 8);
  return r.v;
}

#define WSTRIDE_A 168   // 134 features padded to 5 K-chunks of 32 (bf16 elems)
#define WSTRIDE_B 272   // 256 features = 8 K-chunks + pad

// --------------------------- prep kernels ----------------------------------
__global__ void __launch_bounds__(256)
prep_vf_kernel(const float* __restrict__ src, __bf16* __restrict__ dst, int n8) {
  const int i = blockIdx.x * blockDim.x + threadIdx.x;
  if (i >= n8) return;
  const float4 a = ((const float4*)src)[2 * i];
  const float4 b = ((const float4*)src)[2 * i + 1];
  Bf8 o;
  o.e[0] = f2bf(a.x); o.e[1] = f2bf(a.y); o.e[2] = f2bf(a.z); o.e[3] = f2bf(a.w);
  o.e[4] = f2bf(b.x); o.e[5] = f2bf(b.y); o.e[6] = f2bf(b.z); o.e[7] = f2bf(b.w);
  ((uint4*)dst)[i] = o.q;
}

__global__ void __launch_bounds__(256)
prep_edge_kernel(const int* __restrict__ bond_adj, const int* __restrict__ edge,
                 const float* __restrict__ bond_table,
                 __bf16* __restrict__ dst, int n) {
  const int i = blockIdx.x * blockDim.x + threadIdx.x;
  if (i >= n) return;
  const int code = edge[bond_adj[i]];
  const float* bt = bond_table + code * 6;
  Bf8 o;
#pragma unroll
  for (int e = 0; e < 6; ++e) o.e[e] = f2bf(bt[e]);
  o.e[6] = bfzero(); o.e[7] = bfzero();
  ((uint4*)dst)[i] = o.q;
}

__global__ void __launch_bounds__(256)
prep_wpad_kernel(const float* __restrict__ src, __bf16* __restrict__ dst,
                 int rows, int srcCols, int dstStride) {
  const int i = blockIdx.x * blockDim.x + threadIdx.x;
  if (i >= rows * dstStride) return;
  const int r = i / dstStride, f = i - r * dstStride;
  dst[i] = (f < srcCols) ? f2bf(src[r * srcCols + f]) : bfzero();
}

// ------------------------- kernel A: neighbor GEMM -------------------------
__global__ void __launch_bounds__(256)
wln_nei_kernel(const __bf16* __restrict__ vfb,       // (B*V, 128) bf16
               const int*    __restrict__ atom_adj,  // (B*V*16)
               const __bf16* __restrict__ edge_pack, // (B*V*16, 8) bf16
               const float*  __restrict__ nbs_mask,  // (B*V, 16)
               const __bf16* __restrict__ W2p,       // (128, WSTRIDE_A) bf16
               const float*  __restrict__ b2,        // (128)
               __bf16*       __restrict__ tmpb,      // (B*V, 128) bf16 out
               int totalBV)
{
  extern __shared__ __bf16 wlds[];                   // [128][WSTRIDE_A]

  // stage pre-padded bf16 weights: pure uint4 copy
  {
    const uint4* ws = (const uint4*)W2p;
    uint4* wd = (uint4*)wlds;
    for (int i = threadIdx.x; i < 128 * WSTRIDE_A / 8; i += blockDim.x)
      wd[i] = ws[i];
  }
  __syncthreads();

  const int lane = threadIdx.x & 31;
  const int wid  = threadIdx.x >> 5;
  const int h = lane >> 4;      // lane half
  const int r = lane & 15;      // A row (=k) / B & D column
  const int gw = blockIdx.x * (blockDim.x >> 5) + wid;
  const int nW = gridDim.x * (blockDim.x >> 5);

  for (int bv = gw; bv < totalBV; bv += nW) {
    // A fragments: rows = 16 neighbors; chunks 0..3 vertex, 4 edge.
    // A layout used here (contiguous K; B matches): lane l: M=l&15, e=0..15 -> K=16h+e
    const int ga = atom_adj[bv * 16 + r];
    const __bf16* src = vfb + (size_t)ga * 128;
    BfVec A[5];
#pragma unroll
    for (int c = 0; c < 4; ++c) {
      const int base = 32 * c + 16 * h;
      A[c].q[0] = *(const uint4*)(src + base);
      A[c].q[1] = *(const uint4*)(src + base + 8);
    }
    // unconditional global load (lanes 16..31 read redundantly), select on DATA
    const uint4 z4 = {0, 0, 0, 0};
    const uint4 ep = *(const uint4*)(edge_pack + (size_t)(bv * 16 + r) * 8);
    A[4].q[0] = (h == 0) ? ep : z4;
    A[4].q[1] = z4;

    // this lane's 8 accumulator rows are neighbors m = 8h + j
    const float* mr = nbs_mask + bv * 16 + 8 * h;
    const float4 m0 = *(const float4*)(mr);
    const float4 m1 = *(const float4*)(mr + 4);
    const float mk[8] = {m0.x, m0.y, m0.z, m0.w, m1.x, m1.y, m1.z, m1.w};

#pragma unroll
    for (int nt = 0; nt < 8; ++nt) {
      v8f acc = {};
      // B fragment: lane col n=r, d = nt*16+r, K = 16h + e, f = 32c + K
      const __bf16* wrow = wlds + (nt * 16 + r) * WSTRIDE_A + 16 * h;
#pragma unroll
      for (int c = 0; c < 5; ++c) {
        v16bf Bf = load_bf16x16(wrow + 32 * c);
        acc = __builtin_amdgcn_wmma_f32_16x16x32_bf16(
            false, A[c].v, false, Bf, (short)0, acc, false, false);
      }
      const float bias = b2[nt * 16 + r];
      float s = 0.f;
#pragma unroll
      for (int j = 0; j < 8; ++j) s = fmaf(lrelu(acc[j] + bias), mk[j], s);
      s += __shfl_xor(s, 16, 32);          // combine the two 8-row halves
      if (h == 0) tmpb[bv * 128 + nt * 16 + r] = f2bf(s);
    }
  }
}

// ------------------------- kernel B: update GEMM ---------------------------
__global__ void __launch_bounds__(256)
wln_update_kernel(__bf16*       __restrict__ vfb,   // (B*V,128) bf16, in-place
                  const __bf16* __restrict__ neib,  // (B*V,128) bf16
                  const __bf16* __restrict__ W1p,   // (128, WSTRIDE_B) bf16
                  const float*  __restrict__ b1,    // (128)
                  float*        __restrict__ outf,  // (B*V,128) f32
                  int writeF32)
{
  extern __shared__ __bf16 wlds[];                  // [64][WSTRIDE_B]

  const int lane = threadIdx.x & 31;
  const int wid  = threadIdx.x >> 5;
  const int h = lane >> 4, r = lane & 15;
  const int t = blockIdx.x * (blockDim.x >> 5) + wid;   // exact tile id

  // A fragments: 16 consecutive (b,v) rows; chunks 0..3 = vf, 4..7 = nei
  const __bf16* srcv = vfb  + (size_t)(t * 16 + r) * 128;
  const __bf16* srcn = neib + (size_t)(t * 16 + r) * 128;
  BfVec A[8];
#pragma unroll
  for (int c = 0; c < 4; ++c) {
    const int base = 32 * c + 16 * h;
    A[c].q[0]     = *(const uint4*)(srcv + base);
    A[c].q[1]     = *(const uint4*)(srcv + base + 8);
    A[4 + c].q[0] = *(const uint4*)(srcn + base);
    A[4 + c].q[1] = *(const uint4*)(srcn + base + 8);
  }

  for (int g = 0; g < 2; ++g) {          // two 64-output-dim weight groups
    __syncthreads();
    {
      const uint4* ws = (const uint4*)(W1p + g * 64 * WSTRIDE_B);
      uint4* wd = (uint4*)wlds;
      for (int i = threadIdx.x; i < 64 * WSTRIDE_B / 8; i += blockDim.x)
        wd[i] = ws[i];
    }
    __syncthreads();

#pragma unroll
    for (int ntl = 0; ntl < 4; ++ntl) {
      const int nt = g * 4 + ntl;
      v8f acc = {};
      const __bf16* wrow = wlds + (ntl * 16 + r) * WSTRIDE_B + 16 * h;
#pragma unroll
      for (int c = 0; c < 8; ++c) {
        v16bf Bf = load_bf16x16(wrow + 32 * c);
        acc = __builtin_amdgcn_wmma_f32_16x16x32_bf16(
            false, A[c].v, false, Bf, (short)0, acc, false, false);
      }
      const float bias = b1[nt * 16 + r];
#pragma unroll
      for (int j = 0; j < 8; ++j) {
        const float val = lrelu(acc[j] + bias);
        const size_t ix = (size_t)(t * 16 + j + 8 * h) * 128 + nt * 16 + r;
        vfb[ix] = f2bf(val);               // bf16 mirror for next layer
        if (writeF32) outf[ix] = val;      // f32 only on the final layer
      }
    }
  }
}

// ---------------------------------------------------------------------------
extern "C" void kernel_launch(void* const* d_in, const int* in_sizes, int n_in,
                              void* d_out, int out_size, void* d_ws, size_t ws_size,
                              hipStream_t stream) {
  const float* tgt        = (const float*)d_in[0];   // (64,256,128)
  const int*   edge       = (const int*)  d_in[1];   // (64,512)
  const int*   atom_adj   = (const int*)  d_in[2];   // (262144)
  const int*   bond_adj   = (const int*)  d_in[3];   // (262144)
  const float* nbs_mask   = (const float*)d_in[4];   // (64,256,16)
  // d_in[5] = tgt_key_padding_mask: unused by the reference
  const float* bond_table = (const float*)d_in[6];   // (32,6)
  const float* U2_w       = (const float*)d_in[7];   // (3,128,134)
  const float* U2_b       = (const float*)d_in[8];   // (3,128)
  const float* U1_w       = (const float*)d_in[9];   // (3,128,256)
  const float* U1_b       = (const float*)d_in[10];  // (3,128)

  float* outf = (float*)d_out;                       // (64,256,128)

  const int totalBV = 64 * 256;                      // 16384
  const int NVF = totalBV * 128;                     // 2,097,152 elements

  // d_ws layout (bf16 elements, all 16B aligned)
  __bf16* vfb   = (__bf16*)d_ws;                     // bf16 vf mirror
  __bf16* tmpb  = vfb  + NVF;                        // bf16 nei_label
  __bf16* epack = tmpb + NVF;                        // (262144, 8) edge packets
  __bf16* W2p   = epack + (size_t)totalBV * 16 * 8;  // (3,128,168)
  __bf16* W1p   = W2p + (size_t)3 * 128 * WSTRIDE_A; // (3,128,272)

  const size_t ldsA = (size_t)128 * WSTRIDE_A * sizeof(unsigned short); // 43008 B
  const size_t ldsB = (size_t)64  * WSTRIDE_B * sizeof(unsigned short); // 34816 B

  // one-time conversions (run every launch for determinism; cheap)
  prep_vf_kernel  <<<(NVF / 8 + 255) / 256, 256, 0, stream>>>(tgt, vfb, NVF / 8);
  prep_edge_kernel<<<(totalBV * 16 + 255) / 256, 256, 0, stream>>>(
      bond_adj, edge, bond_table, epack, totalBV * 16);
  prep_wpad_kernel<<<(3 * 128 * WSTRIDE_A + 255) / 256, 256, 0, stream>>>(
      U2_w, W2p, 3 * 128, 134, WSTRIDE_A);
  prep_wpad_kernel<<<(3 * 128 * WSTRIDE_B + 255) / 256, 256, 0, stream>>>(
      U1_w, W1p, 3 * 128, 256, WSTRIDE_B);

  for (int i = 0; i < 3; ++i) {
    wln_nei_kernel<<<512, 256, ldsA, stream>>>(
        vfb, atom_adj, epack, nbs_mask,
        W2p + (size_t)i * 128 * WSTRIDE_A, U2_b + (size_t)i * 128,
        tmpb, totalBV);
    // 1024 tiles of 16 rows; 8 waves/block -> 128 blocks, one tile per wave
    wln_update_kernel<<<128, 256, ldsB, stream>>>(
        vfb, tmpb, W1p + (size_t)i * 128 * WSTRIDE_B, U1_b + (size_t)i * 128,
        outf, (i == 2) ? 1 : 0);
  }
}
